// InvariantPointAttention_18511309046035
// MI455X (gfx1250) — matrix operand order repaired and checked by reading, not compiled, not run on hardware
//
#include <hip/hip_runtime.h>
#include <hip/hip_bf16.h>
#include <stdint.h>

typedef __attribute__((ext_vector_type(16))) __bf16 v16bf;
typedef __attribute__((ext_vector_type(8)))  float  v8f;
typedef __attribute__((ext_vector_type(4)))  unsigned int u32x4;
typedef __attribute__((ext_vector_type(8)))  int  i32x8;
typedef __attribute__((ext_vector_type(4)))  int  i32x4;

#if defined(__has_builtin)
# if __has_builtin(__builtin_amdgcn_tensor_load_to_lds)
#  define HAVE_TDM 1
# endif
#endif
#ifndef HAVE_TDM
# define HAVE_TDM 0
#endif

#define BB   2
#define LL   1024
#define DM   384
#define NH   12
#define HD   32
#define MROWS (BB*LL)      // 2048
#define NCAT 1856          // 384*3 + 192 + 192 + 320 (point blocks padded)
#define OFF_QS 0
#define OFF_KS 384
#define OFF_VS 768
#define OFF_QP 1152
#define OFF_KP 1344
#define OFF_VP 1536
#define FEATD 768

// ---------------- WMMA fragment helpers (CDNA5 16x16x32 bf16 layouts) ----------

// A-matrix 16x32 bf16: lanes 0-15 -> M=lane, K chunks {0..7, 16..23};
// lanes 16-31 -> M=lane-16, K chunks {8..15, 24..31}.
static __device__ __forceinline__ v16bf load_a16(const __bf16* A, int lda,
                                                 int m0, int k0, int lane) {
  int hf  = lane >> 4;
  int row = m0 + (lane & 15);
  const __bf16* p0 = A + (size_t)row * lda + k0 + hf * 8;
  const __bf16* p1 = p0 + 16;
  v16bf r;
#pragma unroll
  for (int i = 0; i < 8; i++) { r[i] = p0[i]; r[i + 8] = p1[i]; }
  return r;
}

// B-matrix 32x16 bf16 (KxN): lane -> N column (lane&15), K chunks {0..7,16..23} / {8..15,24..31} per half.
static __device__ __forceinline__ v16bf load_b16(const __bf16* Bm, int ldb,
                                                 int k0, int n0, int lane) {
  int hf = lane >> 4;
  const __bf16* p = Bm + (size_t)(k0 + hf * 8) * ldb + n0 + (lane & 15);
  v16bf r;
#pragma unroll
  for (int i = 0; i < 8; i++) { r[i] = p[(size_t)i * ldb]; r[i + 8] = p[(size_t)(i + 16) * ldb]; }
  return r;
}

static __device__ __forceinline__ v8f wmma_bf16(v16bf a, v16bf b, v8f c) {
  return __builtin_amdgcn_wmma_f32_16x16x32_bf16(false, a, false, b,
                                                 (short)0, c, false, false);
}

// ---------------- CDNA5 async global -> LDS (ASYNCcnt) --------------------------

static __device__ __forceinline__ void async_b128(uint32_t lds_off, const __bf16* g) {
  // GLOBAL_LOAD_ASYNC_TO_LDS_B128: VDST = LDS byte address, VADDR = 64-bit global
  asm volatile("global_load_async_to_lds_b128 %0, %1, off"
               :: "v"(lds_off), "v"(g) : "memory");
}

template <int N>
static __device__ __forceinline__ void wait_asynccnt() {
  asm volatile("s_wait_asynccnt %0" :: "n"(N) : "memory");
}

template <short N>
static __device__ __forceinline__ void wait_tensorcnt() {
#if HAVE_TDM
  __builtin_amdgcn_s_wait_tensorcnt(N);
#endif
}

// A tile: 16 rows x 32 cols bf16 (1 KB). 2 async B128 per lane (2 ASYNCcnt/wave).
static __device__ __forceinline__ void stage_tileA(uint32_t ldsA, const __bf16* A,
                                                   int lda, int m0, int k0, int lane) {
  int row = lane >> 1, seg = lane & 1;
  const __bf16* g = A + (size_t)(m0 + row) * lda + k0 + seg * 16;
  uint32_t l = ldsA + row * 64 + seg * 32;
  async_b128(l, g);
  async_b128(l + 16, g + 8);
}

// B tile: 32 rows x 64 cols bf16 (4 KB).
// TDM path: one TENSOR_LOAD_TO_LDS per wave with a 2D D# descriptor.
// Fallback: 8 async B128 per lane.
static __device__ __forceinline__ void stage_tileB(uint32_t ldsB, const __bf16* Bm,
                                                   int N, int K, int k0, int n0,
                                                   int lane) {
#if HAVE_TDM
  (void)lane;
  uint64_t ga = (uint64_t)(uintptr_t)(Bm + (size_t)k0 * N + n0);
  u32x4 g0 = {};
  g0[0] = 1u;                                             // count=1, user-mode D#
  g0[1] = ldsB;                                           // lds_addr (bytes)
  g0[2] = (uint32_t)(ga & 0xffffffffu);                   // global_addr[31:0]
  g0[3] = (uint32_t)((ga >> 32) & 0x01ffffffu)            // global_addr[56:32]
        | (2u << 30);                                     // type = 2 ("image")
  uint32_t nd = (uint32_t)N, kd = (uint32_t)K;
  i32x8 g1 = {};
  g1[0] = (int)(1u << 16);                                // data_size = 2 bytes
  g1[1] = (int)((nd & 0xffffu) << 16);                    // tensor_dim0[15:0]
  g1[2] = (int)((nd >> 16) | ((kd & 0xffffu) << 16));     // td0[31:16] | td1[15:0]
  g1[3] = (int)((kd >> 16) | (64u << 16));                // td1[31:16] | tile_dim0=64
  g1[4] = (int)32u;                                       // tile_dim1=32, tile_dim2=0
  g1[5] = (int)nd;                                        // tensor_dim0_stride[31:0]
  g1[6] = 0;                                              // stride hi / dim1 stride lo
  g1[7] = 0;
  i32x4 z4 = {};
# if __clang_major__ >= 23
  i32x8 z8 = {};
  __builtin_amdgcn_tensor_load_to_lds(g0, g1, z4, z4, z8, 0);
# else
  __builtin_amdgcn_tensor_load_to_lds(g0, g1, z4, z4, 0);
# endif
#else
  const __bf16* g = Bm + (size_t)(k0 + lane) * N + n0;
  uint32_t l = ldsB + lane * 128;
#pragma unroll
  for (int i = 0; i < 8; i++) async_b128(l + i * 16, g + i * 8);
#endif
}

// ---------------- prep: pack weights to bf16 with padding -----------------------

__global__ void prep_wcat(const float* Wq, const float* Wk, const float* Wv,
                          const float* Wqp, const float* Wkp, const float* Wvp,
                          __bf16* Wcat) {
  int idx = blockIdx.x * blockDim.x + threadIdx.x;
  if (idx >= DM * NCAT) return;
  int k = idx / NCAT, n = idx % NCAT;
  float v = 0.f;
  if (n < 384)       v = Wq[k * 384 + n];
  else if (n < 768)  v = Wk[k * 384 + (n - 384)];
  else if (n < 1152) v = Wv[k * 384 + (n - 768)];
  else if (n < 1344) { int c = n - 1152; if (c < 144) v = Wqp[k * 144 + c]; }
  else if (n < 1536) { int c = n - 1344; if (c < 144) v = Wkp[k * 144 + c]; }
  else               { int c = n - 1536; if (c < 288) v = Wvp[k * 288 + c]; }
  Wcat[idx] = (__bf16)v;
}

__global__ void prep_bcat(const float* bq, const float* bk, const float* bv,
                          const float* bqp, const float* bkp, const float* bvp,
                          float* bcat) {
  int n = blockIdx.x * blockDim.x + threadIdx.x;
  if (n >= NCAT) return;
  float v = 0.f;
  if (n < 384)       v = bq[n];
  else if (n < 768)  v = bk[n - 384];
  else if (n < 1152) v = bv[n - 768];
  else if (n < 1344) { int c = n - 1152; if (c < 144) v = bqp[c]; }
  else if (n < 1536) { int c = n - 1344; if (c < 144) v = bkp[c]; }
  else               { int c = n - 1536; if (c < 288) v = bvp[c]; }
  bcat[n] = v;
}

__global__ void prep_wo(const float* Wo, __bf16* Wob) {
  int idx = blockIdx.x * blockDim.x + threadIdx.x;
  if (idx >= FEATD * DM) return;
  Wob[idx] = (__bf16)Wo[idx];
}

// ---------------- layernorm -> bf16 h -------------------------------------------

__global__ void __launch_bounds__(128) ln_kernel(const float* s, const float* g,
                                                 const float* be, __bf16* hout) {
  int row = blockIdx.x;
  int t = threadIdx.x;
  __shared__ float red[128];
  const float* sr = s + (size_t)row * DM;
  float x0 = sr[t], x1 = sr[t + 128], x2 = sr[t + 256];
  red[t] = x0 + x1 + x2;
  __syncthreads();
  for (int st = 64; st > 0; st >>= 1) { if (t < st) red[t] += red[t + st]; __syncthreads(); }
  float mu = red[0] * (1.f / DM);
  __syncthreads();
  float d0 = x0 - mu, d1 = x1 - mu, d2 = x2 - mu;
  red[t] = d0 * d0 + d1 * d1 + d2 * d2;
  __syncthreads();
  for (int st = 64; st > 0; st >>= 1) { if (t < st) red[t] += red[t + st]; __syncthreads(); }
  float rstd = rsqrtf(red[0] * (1.f / DM) + 1e-5f);
  __bf16* ho = hout + (size_t)row * DM;
  ho[t]       = (__bf16)(d0 * rstd * g[t]       + be[t]);
  ho[t + 128] = (__bf16)(d1 * rstd * g[t + 128] + be[t + 128]);
  ho[t + 256] = (__bf16)(d2 * rstd * g[t + 256] + be[t + 256]);
}

// ---------------- WMMA GEMM: TDM + async-LDS double-buffered ---------------------
// 4 waves / block; per wave 16x64 tile. A via ASYNCcnt, B via TENSORcnt (TDM).
// mode 0: Cb(bf16) = A@B + bias        mode 1: Cf(f32) = A@B + bias + resid

#define GEMM_WBUF 2560   // elems per (wave,buffer): A 512 + B 2048

__global__ void __launch_bounds__(128) wmma_gemm(const __bf16* A, const __bf16* Bm,
                                                 const float* bias, const float* resid,
                                                 float* Cf, __bf16* Cb,
                                                 int Mr, int N, int K, int mode) {
  __shared__ __bf16 smem[4 * 2 * GEMM_WBUF];   // 40 KB
  int wid  = (blockIdx.x * blockDim.x + threadIdx.x) >> 5;
  int lane = threadIdx.x & 31;
  int wb   = threadIdx.x >> 5;
  int ntw = N >> 6;
  int nw = wid % ntw;
  int mw = wid / ntw;
  if (mw >= (Mr >> 4)) return;     // wave-uniform; no barriers in this kernel
  int m0 = mw << 4, n0 = nw << 6;

  int cbo = wb * 2 * GEMM_WBUF;                        // wave's element base in smem
  uint32_t smem_base = (uint32_t)(uintptr_t)&smem[0];  // LDS byte offset (addr[31:0])
  uint32_t lds0 = smem_base + (uint32_t)cbo * 2u;      // buffer 0 (bytes)
  uint32_t lds1 = lds0 + GEMM_WBUF * 2u;               // buffer 1 (bytes)

  // prologue: stage buffer 0 for k0 = 0
  stage_tileA(lds0, A, K, m0, 0, lane);
  stage_tileB(lds0 + 512 * 2, Bm, N, K, 0, n0, lane);

  v8f acc0 = {}, acc1 = {}, acc2 = {}, acc3 = {};
  int buf = 0;
  for (int k0 = 0; k0 < K; k0 += 32, buf ^= 1) {
    bool more = (k0 + 32 < K);
    if (more) {   // stage next buffer while current computes
      uint32_t lnext = buf ? lds0 : lds1;
      stage_tileA(lnext, A, K, m0, k0 + 32, lane);
      stage_tileB(lnext + 512 * 2, Bm, N, K, k0 + 32, n0, lane);
#if HAVE_TDM
      wait_asynccnt<2>();       // A loads retire in order: current A resident
      wait_tensorcnt<1>();      // TDM in order: current B tile resident
#else
      wait_asynccnt<10>();
#endif
    } else {
      wait_asynccnt<0>();
      wait_tensorcnt<0>();
    }
    if (k0 + 64 < K)            // far-ahead L2 prefetch hint
      __builtin_prefetch(Bm + (size_t)(k0 + 64 + lane) * N + n0, 0, 1);

    // direct integer-offset GEPs on the __shared__ array -> ds_load path
    const __bf16* sA = smem + cbo + (buf ? GEMM_WBUF : 0);
    const __bf16* sB = sA + 512;
    v16bf a  = load_a16(sA, 32, 0, 0, lane);
    v16bf b0 = load_b16(sB, 64, 0, 0,  lane);
    v16bf b1 = load_b16(sB, 64, 0, 16, lane);
    v16bf b2 = load_b16(sB, 64, 0, 32, lane);
    v16bf b3 = load_b16(sB, 64, 0, 48, lane);
    acc0 = wmma_bf16(a, b0, acc0);
    acc1 = wmma_bf16(a, b1, acc1);
    acc2 = wmma_bf16(a, b2, acc2);
    acc3 = wmma_bf16(a, b3, acc3);
  }

  int hf = lane >> 4, cw = lane & 15;
  v8f accs[4] = {acc0, acc1, acc2, acc3};
#pragma unroll
  for (int t2 = 0; t2 < 4; t2++) {
#pragma unroll
    for (int r = 0; r < 8; r++) {
      int rowg = m0 + r + 8 * hf;
      int colg = n0 + 16 * t2 + cw;
      float v = accs[t2][r] + bias[colg];
      if (mode == 0) Cb[(size_t)rowg * N + colg] = (__bf16)v;
      else           Cf[(size_t)rowg * N + colg] = v + resid[(size_t)rowg * N + colg];
    }
  }
}

// ---------------- reorg: scalar q/k/v into WMMA-native layouts -------------------

__global__ void reorg_scalar(const __bf16* proj, __bf16* qs_a, __bf16* ksT, __bf16* vs_b) {
  int idx = blockIdx.x * blockDim.x + threadIdx.x;
  if (idx >= BB * NH * LL * HD) return;
  int d = idx % HD;  int rest = idx / HD;
  int l = rest % LL; rest /= LL;
  int h = rest % NH; int b = rest / NH;
  int row = b * LL + l;
  int bh  = b * NH + h;
  __bf16 q = proj[(size_t)row * NCAT + OFF_QS + h * HD + d];
  __bf16 k = proj[(size_t)row * NCAT + OFF_KS + h * HD + d];
  __bf16 v = proj[(size_t)row * NCAT + OFF_VS + h * HD + d];
  qs_a[((size_t)bh * LL + l) * HD + d] = q;
  ksT [((size_t)bh * HD + d) * LL + l] = k;   // B-layout: [K=32][L]
  vs_b[((size_t)bh * LL + l) * HD + d] = v;   // B-layout: [K=j][32]
}

// ---------------- points: rotate/translate, pad K 12->32 / 24->32, norms ---------

__global__ void points_kernel(const __bf16* proj, const float* trans, const float* rots,
                              __bf16* qg_a, __bf16* kgT, __bf16* vg_b,
                              float* q2, float* k2) {
  int idx = blockIdx.x * blockDim.x + threadIdx.x;
  if (idx >= BB * LL * NH) return;
  int h = idx % NH; int rest = idx / NH;
  int l = rest % LL; int b = rest / LL;
  int row = b * LL + l;
  int bh  = b * NH + h;
  float T[3], R[9];
#pragma unroll
  for (int c = 0; c < 3; c++) T[c] = trans[row * 3 + c];
#pragma unroll
  for (int e = 0; e < 9; e++) R[e] = rots[row * 9 + e];

  float qg[12], kg[12];
  float q2s = 0.f, k2s = 0.f;
#pragma unroll
  for (int p = 0; p < 4; p++) {
    float qp[3], kp[3];
#pragma unroll
    for (int c = 0; c < 3; c++) {
      qp[c] = (float)proj[(size_t)row * NCAT + OFF_QP + h * 12 + p * 3 + c];
      kp[c] = (float)proj[(size_t)row * NCAT + OFF_KP + h * 12 + p * 3 + c];
    }
#pragma unroll
    for (int d = 0; d < 3; d++) {
      float qv = qp[0] * R[0 * 3 + d] + qp[1] * R[1 * 3 + d] + qp[2] * R[2 * 3 + d] + T[d];
      float kv = kp[0] * R[0 * 3 + d] + kp[1] * R[1 * 3 + d] + kp[2] * R[2 * 3 + d] + T[d];
      qg[p * 3 + d] = qv; kg[p * 3 + d] = kv;
      q2s += qv * qv; k2s += kv * kv;
    }
  }
  q2[(size_t)bh * LL + l] = q2s;
  k2[(size_t)bh * LL + l] = k2s;
  __bf16* qa = qg_a + ((size_t)bh * LL + l) * HD;
#pragma unroll
  for (int k = 0; k < HD; k++) qa[k] = (__bf16)(k < 12 ? qg[k] : 0.f);
#pragma unroll
  for (int k = 0; k < HD; k++)
    kgT[((size_t)bh * HD + k) * LL + l] = (__bf16)(k < 12 ? kg[k] : 0.f);

  __bf16* vb = vg_b + ((size_t)bh * LL + l) * HD;
#pragma unroll
  for (int p = 0; p < 8; p++) {
    float vp[3];
#pragma unroll
    for (int c = 0; c < 3; c++)
      vp[c] = (float)proj[(size_t)row * NCAT + OFF_VP + h * 24 + p * 3 + c];
#pragma unroll
    for (int d = 0; d < 3; d++) {
      float vv = vp[0] * R[0 * 3 + d] + vp[1] * R[1 * 3 + d] + vp[2] * R[2 * 3 + d] + T[d];
      vb[p * 3 + d] = (__bf16)vv;
    }
  }
#pragma unroll
  for (int k = 24; k < HD; k++) vb[k] = (__bf16)0.f;
}

// ---------------- flash IPA attention: 1 wave per (b,h, 16-row i-tile) -----------

__global__ void __launch_bounds__(256) attn_kernel(
    const __bf16* qs_a, const __bf16* ksT, const __bf16* vs_b,
    const __bf16* qg_a, const __bf16* kgT, const __bf16* vg_b,
    const float* q2, const float* k2, const uint8_t* mask,
    const float* head_w, float* res_s, float* res_p) {
  __shared__ __bf16 pbuf[8][16 * 32];
  int w    = threadIdx.x >> 5;
  int lane = threadIdx.x & 31;
  int wid  = blockIdx.x * 8 + w;
  int itile = wid & 63;          // L/16 = 64 i-tiles
  int bh    = wid >> 6;
  int b = bh / NH, h = bh % NH;
  int i0 = itile << 4;
  int hf = lane >> 4, col = lane & 15;

  float w_h = logf(1.f + __expf(head_w[h]));   // softplus
  const __bf16* qsb = qs_a + (size_t)bh * LL * HD;
  const __bf16* qgb = qg_a + (size_t)bh * LL * HD;
  const __bf16* ktb = ksT  + (size_t)bh * HD * LL;
  const __bf16* kgb = kgT  + (size_t)bh * HD * LL;
  const __bf16* vsb = vs_b + (size_t)bh * LL * HD;
  const __bf16* vgb = vg_b + (size_t)bh * LL * HD;

  v16bf aqs = load_a16(qsb, HD, i0, 0, lane);
  v16bf aqg = load_a16(qgb, HD, i0, 0, lane);
  float q2r[8];
#pragma unroll
  for (int r = 0; r < 8; r++) q2r[r] = q2[(size_t)bh * LL + i0 + r + 8 * hf];

  float m[8], ssum[8];
#pragma unroll
  for (int r = 0; r < 8; r++) { m[r] = -1e30f; ssum[r] = 0.f; }
  v8f os0 = {}, os1 = {}, og0 = {}, og1 = {};
  const float isq = 0.17677669529663687f;   // 1/sqrt(32)
  v8f z = {};

  for (int j0 = 0; j0 < LL; j0 += 32) {
    // logits: scalar attn (K=32) + point cross (K padded 12->32)
    v16bf bks0 = load_b16(ktb, LL, 0, j0,      lane);
    v16bf bks1 = load_b16(ktb, LL, 0, j0 + 16, lane);
    v16bf bkg0 = load_b16(kgb, LL, 0, j0,      lane);
    v16bf bkg1 = load_b16(kgb, LL, 0, j0 + 16, lane);
    v8f s0 = wmma_bf16(aqs, bks0, z);
    v8f s1 = wmma_bf16(aqs, bks1, z);
    v8f c0 = wmma_bf16(aqg, bkg0, z);
    v8f c1 = wmma_bf16(aqg, bkg1, z);

    float k2a = k2[(size_t)bh * LL + j0 + col];
    float k2b = k2[(size_t)bh * LL + j0 + 16 + col];
    int mk0 = mask[b * LL + j0 + col];
    int mk1 = mask[b * LL + j0 + 16 + col];

#pragma unroll
    for (int r = 0; r < 8; r++) {
      float a0 = s0[r] * isq - 0.5f * w_h * (q2r[r] + k2a - 2.f * c0[r]);
      float a1 = s1[r] * isq - 0.5f * w_h * (q2r[r] + k2b - 2.f * c1[r]);
      a0 = mk0 ? a0 : -1e30f;
      a1 = mk1 ? a1 : -1e30f;
      // row max across the 16 lanes of this half (C-layout row lives there)
      float mr = fmaxf(a0, a1);
      mr = fmaxf(mr, __shfl_xor(mr, 1));
      mr = fmaxf(mr, __shfl_xor(mr, 2));
      mr = fmaxf(mr, __shfl_xor(mr, 4));
      mr = fmaxf(mr, __shfl_xor(mr, 8));
      float mn = fmaxf(m[r], mr);
      float sc = __expf(m[r] - mn);
      float p0 = __expf(a0 - mn);
      float p1 = __expf(a1 - mn);
      float rs = p0 + p1;
      rs += __shfl_xor(rs, 1); rs += __shfl_xor(rs, 2);
      rs += __shfl_xor(rs, 4); rs += __shfl_xor(rs, 8);
      ssum[r] = ssum[r] * sc + rs;
      m[r] = mn;
      os0[r] *= sc; os1[r] *= sc; og0[r] *= sc; og1[r] *= sc;
      pbuf[w][(r + 8 * hf) * 32 + col]      = (__bf16)p0;
      pbuf[w][(r + 8 * hf) * 32 + col + 16] = (__bf16)p1;
    }
    __syncthreads();              // C-layout -> A-layout via LDS
    v16bf ap  = load_a16(&pbuf[w][0], 32, 0, 0, lane);
    v16bf bv0 = load_b16(vsb, HD, j0, 0,  lane);
    v16bf bv1 = load_b16(vsb, HD, j0, 16, lane);
    v16bf bg0 = load_b16(vgb, HD, j0, 0,  lane);
    v16bf bg1 = load_b16(vgb, HD, j0, 16, lane);
    os0 = wmma_bf16(ap, bv0, os0);
    os1 = wmma_bf16(ap, bv1, os1);
    og0 = wmma_bf16(ap, bg0, og0);
    og1 = wmma_bf16(ap, bg1, og1);
    __syncthreads();
  }

#pragma unroll
  for (int r = 0; r < 8; r++) {
    float inv = 1.f / ssum[r];
    int irow = i0 + r + 8 * hf;
    res_s[((size_t)bh * LL + irow) * HD + col]      = os0[r] * inv;
    res_s[((size_t)bh * LL + irow) * HD + col + 16] = os1[r] * inv;
    res_p[((size_t)bh * LL + irow) * 24 + col]      = og0[r] * inv;
    if (col < 8)
      res_p[((size_t)bh * LL + irow) * 24 + 16 + col] = og1[r] * inv;
  }
}

// ---------------- post: inverse frame, norms, build 768-wide feature -------------

__global__ void post_kernel(const float* res_s, const float* res_p,
                            const float* trans, const float* rots, __bf16* feat) {
  int idx = blockIdx.x * blockDim.x + threadIdx.x;
  if (idx >= BB * LL * NH) return;
  int h = idx % NH; int rest = idx / NH;
  int l = rest % LL; int b = rest / LL;
  int row = b * LL + l;
  int bh  = b * NH + h;
  __bf16* fr = feat + (size_t)row * FEATD;
  const float* rs = res_s + ((size_t)bh * LL + l) * HD;
#pragma unroll
  for (int d = 0; d < HD; d++) fr[h * HD + d] = (__bf16)rs[d];
  float T[3], R[9];
#pragma unroll
  for (int c = 0; c < 3; c++) T[c] = trans[row * 3 + c];
#pragma unroll
  for (int e = 0; e < 9; e++) R[e] = rots[row * 9 + e];
  const float* rp = res_p + ((size_t)bh * LL + l) * 24;
#pragma unroll
  for (int p = 0; p < 8; p++) {
    float df[3];
#pragma unroll
    for (int c = 0; c < 3; c++) df[c] = rp[p * 3 + c] - T[c];
    float n2 = 0.f;
#pragma unroll
    for (int d = 0; d < 3; d++) {
      float lv = df[0] * R[d * 3 + 0] + df[1] * R[d * 3 + 1] + df[2] * R[d * 3 + 2];
      fr[384 + h * 24 + p * 3 + d] = (__bf16)lv;
      n2 += lv * lv;
    }
    fr[672 + h * 8 + p] = (__bf16)sqrtf(n2);
  }
}

// ---------------- launch ---------------------------------------------------------

extern "C" void kernel_launch(void* const* d_in, const int* in_sizes, int n_in,
                              void* d_out, int out_size, void* d_ws, size_t ws_size,
                              hipStream_t stream) {
  const float* s     = (const float*)d_in[0];
  const float* trans = (const float*)d_in[1];
  const float* rots  = (const float*)d_in[2];
  const uint8_t* mask = (const uint8_t*)d_in[3];
  const float* ln_g  = (const float*)d_in[4];
  const float* ln_b  = (const float*)d_in[5];
  const float* Wq = (const float*)d_in[6],  *bq = (const float*)d_in[7];
  const float* Wk = (const float*)d_in[8],  *bk = (const float*)d_in[9];
  const float* Wv = (const float*)d_in[10], *bv = (const float*)d_in[11];
  const float* Wqp = (const float*)d_in[12], *bqp = (const float*)d_in[13];
  const float* Wkp = (const float*)d_in[14], *bkp = (const float*)d_in[15];
  const float* Wvp = (const float*)d_in[16], *bvp = (const float*)d_in[17];
  const float* head_w = (const float*)d_in[18];
  const float* Wo = (const float*)d_in[19], *bo = (const float*)d_in[20];
  float* out = (float*)d_out;

  char* p = (char*)d_ws;
  auto carve = [&](size_t bytes) -> char* {
    char* r = p; p += (bytes + 255) & ~(size_t)255; return r;
  };
  size_t hdsz = (size_t)BB * NH * LL * HD * 2;
  __bf16* Wcat = (__bf16*)carve((size_t)DM * NCAT * 2);
  float*  bcat = (float*) carve((size_t)NCAT * 4);
  __bf16* Wob  = (__bf16*)carve((size_t)FEATD * DM * 2);
  __bf16* hbf  = (__bf16*)carve((size_t)MROWS * DM * 2);
  __bf16* proj = (__bf16*)carve((size_t)MROWS * NCAT * 2);
  __bf16* qs_a = (__bf16*)carve(hdsz);
  __bf16* ksT  = (__bf16*)carve(hdsz);
  __bf16* vs_b = (__bf16*)carve(hdsz);
  __bf16* qg_a = (__bf16*)carve(hdsz);
  __bf16* kgT  = (__bf16*)carve(hdsz);
  __bf16* vg_b = (__bf16*)carve(hdsz);
  float*  q2   = (float*) carve((size_t)BB * NH * LL * 4);
  float*  k2   = (float*) carve((size_t)BB * NH * LL * 4);
  float*  res_s = (float*)carve((size_t)BB * NH * LL * HD * 4);
  float*  res_p = (float*)carve((size_t)BB * NH * LL * 24 * 4);
  __bf16* feat = (__bf16*)carve((size_t)MROWS * FEATD * 2);
  (void)ws_size; (void)in_sizes; (void)n_in; (void)out_size;

  prep_wcat<<<(DM * NCAT + 255) / 256, 256, 0, stream>>>(Wq, Wk, Wv, Wqp, Wkp, Wvp, Wcat);
  prep_bcat<<<(NCAT + 255) / 256, 256, 0, stream>>>(bq, bk, bv, bqp, bkp, bvp, bcat);
  prep_wo<<<(FEATD * DM + 255) / 256, 256, 0, stream>>>(Wo, Wob);
  ln_kernel<<<MROWS, 128, 0, stream>>>(s, ln_g, ln_b, hbf);
  // proj GEMM: (2048/16)*(1856/64) = 3712 waves -> 928 blocks of 4 waves
  wmma_gemm<<<928, 128, 0, stream>>>(hbf, Wcat, bcat, nullptr, nullptr, proj,
                                     MROWS, NCAT, DM, 0);
  reorg_scalar<<<(BB * NH * LL * HD + 255) / 256, 256, 0, stream>>>(proj, qs_a, ksT, vs_b);
  points_kernel<<<(BB * LL * NH + 255) / 256, 256, 0, stream>>>(proj, trans, rots,
                                                                qg_a, kgT, vg_b, q2, k2);
  // attention: BH * (L/16) = 24*64 = 1536 waves -> 192 blocks of 8 waves
  attn_kernel<<<192, 256, 0, stream>>>(qs_a, ksT, vs_b, qg_a, kgT, vg_b,
                                       q2, k2, mask, head_w, res_s, res_p);
  post_kernel<<<(BB * LL * NH + 255) / 256, 256, 0, stream>>>(res_s, res_p, trans, rots, feat);
  // output GEMM: (2048/16)*(384/64) = 768 waves -> 192 blocks of 4 waves
  wmma_gemm<<<192, 128, 0, stream>>>(feat, Wob, bo, s, out, nullptr,
                                     MROWS, DM, FEATD, 1);
}
